// TransformerBlock_1881195675706
// MI455X (gfx1250) — matrix operand order repaired and checked by reading, not compiled, not run on hardware
//
#include <hip/hip_runtime.h>
#include <hip/hip_bf16.h>
#include <math.h>

// Problem constants (match reference)
#define BB   2
#define SS   2048
#define DD   1024
#define HH   16
#define DH   64
#define DFF  4096
#define TOK  (BB * SS)   // 4096 tokens

typedef __attribute__((ext_vector_type(16))) __bf16 v16bf;
typedef __attribute__((ext_vector_type(8)))  float  v8f;
typedef __attribute__((ext_vector_type(4)))  int    v4i;

union BFrag {
    unsigned short u[16];
    unsigned int   w[8];
    v16bf v;
};

__device__ __forceinline__ unsigned short f2bf(float f) {
    unsigned int x = __float_as_uint(f);
    unsigned int r = x + 0x7FFFu + ((x >> 16) & 1u);   // round-to-nearest-even
    return (unsigned short)(r >> 16);
}

// ---------------------------------------------------------------------------
// Async global -> LDS (CDNA5 GLOBAL_LOAD_ASYNC_TO_LDS_B128, ASYNCcnt-tracked).
// Builtin signature (from hipcc diagnostic): (v4i AS1* src, v4i AS3* dst, Ii, Ii)
// Falls back to a synchronous copy if the builtin is unavailable.
// ---------------------------------------------------------------------------
#if __has_builtin(__builtin_amdgcn_global_load_async_to_lds_b128)
#define HAVE_ASYNC_LDS 1
#else
#define HAVE_ASYNC_LDS 0
#endif

__device__ __forceinline__ void g2l_b128(const unsigned short* __restrict__ g,
                                         unsigned short* l) {
#if HAVE_ASYNC_LDS
    __builtin_amdgcn_global_load_async_to_lds_b128(
        (__attribute__((address_space(1))) v4i*)g,
        (__attribute__((address_space(3))) v4i*)l, 0, 0);
#else
    *(uint4*)l = *(const uint4*)g;
#endif
}

__device__ __forceinline__ void async_wait0() {
#if HAVE_ASYNC_LDS
#if __has_builtin(__builtin_amdgcn_s_wait_asynccnt)
    __builtin_amdgcn_s_wait_asynccnt(0);
#else
    asm volatile("s_wait_asynccnt 0x0" ::: "memory");
#endif
#endif
}

// ---------------------------------------------------------------------------
// fp32 -> bf16 weight conversion (one pass; GEMMs then stream bf16)
// ---------------------------------------------------------------------------
__global__ void cvt_f32_bf16(const float* __restrict__ src,
                             unsigned short* __restrict__ dst, int n) {
    int i = blockIdx.x * blockDim.x + threadIdx.x;
    if (i < n) dst[i] = f2bf(src[i]);
}

// ---------------------------------------------------------------------------
// LayerNorm over D=1024, fp32 in -> bf16 out.  One block (256 thr) per token.
// ---------------------------------------------------------------------------
__global__ __launch_bounds__(256) void ln_bf16(const float* __restrict__ x,
                                               const float* __restrict__ g,
                                               const float* __restrict__ b,
                                               unsigned short* __restrict__ out) {
    __shared__ float s1[256], s2[256];
    const int t = threadIdx.x;
    const size_t row = blockIdx.x;
    const float* xr = x + row * DD;
    float v0 = xr[t], v1 = xr[t + 256], v2 = xr[t + 512], v3 = xr[t + 768];
    s1[t] = v0 + v1 + v2 + v3;
    s2[t] = v0 * v0 + v1 * v1 + v2 * v2 + v3 * v3;
    __syncthreads();
    for (int o = 128; o > 0; o >>= 1) {
        if (t < o) { s1[t] += s1[t + o]; s2[t] += s2[t + o]; }
        __syncthreads();
    }
    float mean = s1[0] * (1.0f / DD);
    float var  = s2[0] * (1.0f / DD) - mean * mean;
    float rinv = rsqrtf(var + 1e-5f);
    unsigned short* orow = out + row * DD;
    orow[t      ] = f2bf((v0 - mean) * rinv * g[t      ] + b[t      ]);
    orow[t + 256] = f2bf((v1 - mean) * rinv * g[t + 256] + b[t + 256]);
    orow[t + 512] = f2bf((v2 - mean) * rinv * g[t + 512] + b[t + 512]);
    orow[t + 768] = f2bf((v3 - mean) * rinv * g[t + 768] + b[t + 768]);
}

// ---------------------------------------------------------------------------
// bf16 WMMA GEMM:  C[M,N] = A[M,K] (bf16, row-major) * W[N,K]^T (bf16) + bias
// Block tile 128x128, 8 waves, each wave 32x64 (2x4 WMMA accumulators).
// Double-buffered LDS with async global->LDS staging.
// mode 0: scatter into q/k/v [B,H,S,DH] bf16      (QKV projection)
// mode 1: outF = acc + bias + resid (fp32)        (O projection + residual)
// mode 2: outB = gelu(acc + bias) (bf16)          (MLP up + exact-erf GELU)
// mode 3: outF = acc + bias + resid (fp32)        (MLP down + residual)
// ---------------------------------------------------------------------------
__global__ __launch_bounds__(256) void gemm_bf16(
    const unsigned short* __restrict__ A, const unsigned short* __restrict__ W,
    const float* __restrict__ bias, int M, int N, int K, int mode,
    float* __restrict__ outF, const float* __restrict__ resid,
    unsigned short* __restrict__ outq, unsigned short* __restrict__ outk,
    unsigned short* __restrict__ outv, unsigned short* __restrict__ outB) {
    __shared__ __align__(16) unsigned short lA[2][128 * 32];
    __shared__ __align__(16) unsigned short lB[2][128 * 32];

    const int t    = threadIdx.x;
    const int lane = t & 31;
    const int wave = t >> 5;
    const int wm   = wave >> 1;      // 0..3 : M sub-tile (32 rows)
    const int wn   = wave & 1;       // 0..1 : N sub-tile (64 cols)
    const int l16  = lane & 15;
    const int hi   = lane >> 4;      // lane half (wave32 WMMA layout)
    const int m0   = blockIdx.y * 128;
    const int n0   = blockIdx.x * 128;

    v8f acc[2][4];
    for (int i = 0; i < 2; i++)
        for (int j = 0; j < 4; j++)
            for (int e = 0; e < 8; e++) acc[i][j][e] = 0.0f;

    // Each tile is 128 rows x 32 k of bf16 = 512 x 16B chunks; 2 chunks/thread.
    auto issue_tiles = [&](int buf, int kk) {
        for (int c = t; c < 512; c += 256) {
            int row = c >> 2, part = c & 3;
            g2l_b128(A + (size_t)(m0 + row) * K + kk + part * 8,
                     &lA[buf][row * 32 + part * 8]);
            g2l_b128(W + (size_t)(n0 + row) * K + kk + part * 8,
                     &lB[buf][row * 32 + part * 8]);
        }
    };

    const int nk = K / 32;
    issue_tiles(0, 0);
    async_wait0();
    __syncthreads();

    for (int ik = 0; ik < nk; ik++) {
        const int cur = ik & 1;
        if (ik + 1 < nk) issue_tiles(cur ^ 1, (ik + 1) * 32);   // prefetch next

        // Gather WMMA fragments, contiguous k order (k = 16*hi + s) in both A and B
        BFrag aF[2], bF[4];
        for (int s = 0; s < 2; s++) {
            int row  = wm * 32 + s * 16 + l16;
            int base = row * 32 + hi * 16;
            for (int h = 0; h < 4; h++) {
                aF[s].w[h]     = *(const unsigned int*)(&lA[cur][base + 2 * h]);
                aF[s].w[4 + h] = *(const unsigned int*)(&lA[cur][base + 8 + 2 * h]);
            }
        }
        for (int s = 0; s < 4; s++) {
            int n    = wn * 64 + s * 16 + l16;
            int base = n * 32 + hi * 16;
            for (int h = 0; h < 8; h++)
                bF[s].w[h] = *(const unsigned int*)(&lB[cur][base + 2 * h]);
        }
        for (int i = 0; i < 2; i++)
            for (int j = 0; j < 4; j++)
                acc[i][j] = __builtin_amdgcn_wmma_f32_16x16x32_bf16(
                    false, aF[i].v, false, bF[j].v, (short)0, acc[i][j], false, false);

        async_wait0();          // next tile landed in LDS
        __syncthreads();        // all waves done reading cur / writing next
    }

    // Epilogue (C layout: lane<16 -> M=r, else M=8+r; N = lane%16)
    for (int i = 0; i < 2; i++)
        for (int j = 0; j < 4; j++)
            for (int r = 0; r < 8; r++) {
                int row = m0 + wm * 32 + i * 16 + (hi ? 8 + r : r);
                int col = n0 + wn * 64 + j * 16 + l16;
                float v = acc[i][j][r] + bias[col];
                size_t idx = (size_t)row * N + col;
                if (mode == 0) {
                    int head = col / (3 * DH), rest = col % (3 * DH);
                    int which = rest / DH, dh = rest % DH;
                    int bb = row / SS, sq = row % SS;
                    size_t di = (((size_t)bb * HH + head) * SS + sq) * DH + dh;
                    unsigned short bv = f2bf(v);
                    if (which == 0)      outq[di] = bv;
                    else if (which == 1) outk[di] = bv;
                    else                 outv[di] = bv;
                } else if (mode == 2) {
                    float gg = 0.5f * v * (1.0f + erff(v * 0.70710678118654752f));
                    outB[idx] = f2bf(gg);
                } else {
                    outF[idx] = v + resid[idx];
                }
            }
}

// ---------------------------------------------------------------------------
// Flash attention with analytic ALiBi + causal mask.
// Grid (S/128, H, B); 256 threads = 8 waves; each wave owns 16 q-rows.
// Both Q·K^T and P·V run as bf16 WMMA with fp32 accumulation.
// ---------------------------------------------------------------------------
__global__ __launch_bounds__(256) void attn_alibi(
    const unsigned short* __restrict__ Q, const unsigned short* __restrict__ Kd,
    const unsigned short* __restrict__ V, unsigned short* __restrict__ O) {
    __shared__ __align__(16) unsigned short lQ[128 * 64];   // reused as per-wave P
    __shared__ __align__(16) unsigned short lK[64 * 64];
    __shared__ __align__(16) unsigned short lV[64 * 64];

    const int t = threadIdx.x, lane = t & 31, wave = t >> 5;
    const int l16 = lane & 15, hi = lane >> 4;
    const int q0 = blockIdx.x * 128;
    const int h  = blockIdx.y, b = blockIdx.z;
    const size_t base = ((size_t)(b * HH + h)) * SS * DH;
    const float slope = exp2f(-0.5f * (float)(h + 1));      // 2^(-8*(h+1)/H)

    // Load Q tile (128 x 64 bf16 = 16 KB) via async path
    for (int c = t; c < 1024; c += 256) {
        int row = c >> 3, part = c & 7;
        g2l_b128(Q + base + (size_t)(q0 + row) * DH + part * 8,
                 &lQ[row * 64 + part * 8]);
    }
    async_wait0();
    __syncthreads();

    // Gather this wave's Q fragments (A layout), rows = wave*16 .. +15
    BFrag aQ[2];
    {
        int rowl = wave * 16 + l16;
        for (int kc = 0; kc < 2; kc++) {
            int kb = kc * 32 + hi * 16;
            for (int hh = 0; hh < 4; hh++) {
                aQ[kc].w[hh]     = *(const unsigned int*)(&lQ[rowl * 64 + kb + 2 * hh]);
                aQ[kc].w[4 + hh] = *(const unsigned int*)(&lQ[rowl * 64 + kb + 8 + 2 * hh]);
            }
        }
    }

    v8f o[4];
    float mrow[8], lrow[8];
    for (int n = 0; n < 4; n++)
        for (int e = 0; e < 8; e++) o[n][e] = 0.0f;
    for (int r = 0; r < 8; r++) { mrow[r] = -3.0e38f; lrow[r] = 0.0f; }

    const int nkt = 2 * blockIdx.x + 2;                 // causal tile count
    for (int kt = 0; kt < nkt; kt++) {
        __syncthreads();                                // prior tile reads done
        const int key0 = kt * 64;
        for (int c = t; c < 512; c += 256) {            // K,V tiles 64x64 bf16
            int row = c >> 3, part = c & 7;
            g2l_b128(Kd + base + (size_t)(key0 + row) * DH + part * 8,
                     &lK[row * 64 + part * 8]);
            g2l_b128(V + base + (size_t)(key0 + row) * DH + part * 8,
                     &lV[row * 64 + part * 8]);
        }
        async_wait0();
        __syncthreads();

        // S = Q K^T  (16 x 64), K-dim = dh (contiguous in lK rows)
        v8f s[4];
        for (int n = 0; n < 4; n++)
            for (int e = 0; e < 8; e++) s[n][e] = 0.0f;
        for (int kc = 0; kc < 2; kc++)
            for (int n = 0; n < 4; n++) {
                BFrag bK;
                int key = n * 16 + l16;
                int kb  = kc * 32 + hi * 16;
                for (int hh = 0; hh < 8; hh++)
                    bK.w[hh] = *(const unsigned int*)(&lK[key * 64 + kb + 2 * hh]);
                s[n] = __builtin_amdgcn_wmma_f32_16x16x32_bf16(
                    false, aQ[kc].v, false, bK.v, (short)0, s[n], false, false);
            }

        // scale + ALiBi + causal mask; per-row tile max
        float tmax[8];
        for (int r = 0; r < 8; r++) tmax[r] = -3.0e38f;
        for (int n = 0; n < 4; n++) {
            int key = key0 + n * 16 + l16;
            for (int r = 0; r < 8; r++) {
                int qr = q0 + wave * 16 + (hi ? 8 + r : r);
                float v = s[n][r] * 0.125f;             // 1/sqrt(64)
                v = (key <= qr) ? (v + slope * (float)(key - qr)) : -1.0e9f;
                s[n][r] = v;
                tmax[r] = fmaxf(tmax[r], v);
            }
        }
        for (int off = 1; off < 16; off <<= 1)          // reduce within halves
            for (int r = 0; r < 8; r++)
                tmax[r] = fmaxf(tmax[r], __shfl_xor(tmax[r], off, 32));

        float scl[8], tsum[8];
        for (int r = 0; r < 8; r++) {
            float mnew = fmaxf(mrow[r], tmax[r]);
            scl[r]  = expf(mrow[r] - mnew);
            mrow[r] = mnew;
            tsum[r] = 0.0f;
        }
        for (int n = 0; n < 4; n++)
            for (int r = 0; r < 8; r++) {
                float p = expf(s[n][r] - mrow[r]);
                s[n][r] = p;
                tsum[r] += p;
            }
        for (int off = 1; off < 16; off <<= 1)
            for (int r = 0; r < 8; r++)
                tsum[r] += __shfl_xor(tsum[r], off, 32);
        for (int r = 0; r < 8; r++) lrow[r] = lrow[r] * scl[r] + tsum[r];
        for (int n = 0; n < 4; n++)
            for (int r = 0; r < 8; r++) o[n][r] *= scl[r];

        // P (C layout) -> this wave's private LDS slice -> A-layout frags
        unsigned short* lP = &lQ[wave * 16 * 64];
        for (int n = 0; n < 4; n++) {
            int col = n * 16 + l16;
            for (int r = 0; r < 8; r++)
                lP[(hi ? 8 + r : r) * 64 + col] = f2bf(s[n][r]);
        }
        // O += P V  (K-dim = key index -> strided gathers for B=V)
        for (int kc = 0; kc < 2; kc++) {
            BFrag aP;
            int kb = kc * 32 + hi * 16;
            for (int hh = 0; hh < 4; hh++) {
                aP.w[hh]     = *(const unsigned int*)(&lP[l16 * 64 + kb + 2 * hh]);
                aP.w[4 + hh] = *(const unsigned int*)(&lP[l16 * 64 + kb + 8 + 2 * hh]);
            }
            for (int n = 0; n < 4; n++) {
                BFrag bV;
                int dh = n * 16 + l16;
                int k0 = kc * 32 + hi * 16;
                for (int j = 0; j < 16; j++)
                    bV.u[j] = lV[(k0 + j) * 64 + dh];
                o[n] = __builtin_amdgcn_wmma_f32_16x16x32_bf16(
                    false, aP.v, false, bV.v, (short)0, o[n], false, false);
            }
        }
    }

    // Normalize and write O (bf16, [B,S,D] with head-concatenated DH)
    for (int n = 0; n < 4; n++) {
        int dh = n * 16 + l16;
        for (int r = 0; r < 8; r++) {
            int qr = q0 + wave * 16 + (hi ? 8 + r : r);
            size_t oi = ((size_t)(b * SS + qr)) * DD + h * DH + dh;
            O[oi] = f2bf(o[n][r] / lrow[r]);
        }
    }
}

// ---------------------------------------------------------------------------
extern "C" void kernel_launch(void* const* d_in, const int* in_sizes, int n_in,
                              void* d_out, int out_size, void* d_ws, size_t ws_size,
                              hipStream_t stream) {
    (void)in_sizes; (void)n_in; (void)out_size; (void)ws_size;
    const float* x     = (const float*)d_in[0];
    /* d_in[1] = alibi tensor (256 MB) — recomputed analytically, never read */
    const float* Wqkv  = (const float*)d_in[2];
    const float* bqkv  = (const float*)d_in[3];
    const float* Wo    = (const float*)d_in[4];
    const float* bo    = (const float*)d_in[5];
    const float* Wup   = (const float*)d_in[6];
    const float* bup   = (const float*)d_in[7];
    const float* Wdown = (const float*)d_in[8];
    const float* bdown = (const float*)d_in[9];
    const float* g1    = (const float*)d_in[10];
    const float* b1    = (const float*)d_in[11];
    const float* g2    = (const float*)d_in[12];
    const float* b2    = (const float*)d_in[13];
    float* out = (float*)d_out;

    char* ws = (char*)d_ws;
    size_t off = 0;
    auto alloc = [&](size_t bytes) -> char* {
        char* p = ws + off;
        off += (bytes + 255) & ~(size_t)255;
        return p;
    };
    unsigned short* wqkv_b = (unsigned short*)alloc((size_t)3 * DD * DD * 2);
    unsigned short* wo_b   = (unsigned short*)alloc((size_t)DD * DD * 2);
    unsigned short* wup_b  = (unsigned short*)alloc((size_t)DFF * DD * 2);
    unsigned short* wdn_b  = (unsigned short*)alloc((size_t)DD * DFF * 2);
    unsigned short* h1     = (unsigned short*)alloc((size_t)TOK * DD * 2);
    unsigned short* qb     = (unsigned short*)alloc((size_t)TOK * DD * 2);
    unsigned short* kb     = (unsigned short*)alloc((size_t)TOK * DD * 2);
    unsigned short* vb     = (unsigned short*)alloc((size_t)TOK * DD * 2);
    unsigned short* ob     = (unsigned short*)alloc((size_t)TOK * DD * 2);
    float*          attn_f = (float*)alloc((size_t)TOK * DD * 4);
    unsigned short* h2     = (unsigned short*)alloc((size_t)TOK * DD * 2);
    unsigned short* hid    = (unsigned short*)alloc((size_t)TOK * DFF * 2);

    int n;
    n = 3 * DD * DD;  cvt_f32_bf16<<<(n + 255) / 256, 256, 0, stream>>>(Wqkv,  wqkv_b, n);
    n = DD * DD;      cvt_f32_bf16<<<(n + 255) / 256, 256, 0, stream>>>(Wo,    wo_b,   n);
    n = DFF * DD;     cvt_f32_bf16<<<(n + 255) / 256, 256, 0, stream>>>(Wup,   wup_b,  n);
    n = DD * DFF;     cvt_f32_bf16<<<(n + 255) / 256, 256, 0, stream>>>(Wdown, wdn_b,  n);

    // LN1 -> bf16
    ln_bf16<<<TOK, 256, 0, stream>>>(x, g1, b1, h1);
    // QKV projection (scatter into [B,H,S,DH])
    gemm_bf16<<<dim3((3 * DD) / 128, TOK / 128), 256, 0, stream>>>(
        h1, wqkv_b, bqkv, TOK, 3 * DD, DD, 0,
        nullptr, nullptr, qb, kb, vb, nullptr);
    // Flash attention with analytic ALiBi
    attn_alibi<<<dim3(SS / 128, HH, BB), 256, 0, stream>>>(qb, kb, vb, ob);
    // O projection + residual (fp32)
    gemm_bf16<<<dim3(DD / 128, TOK / 128), 256, 0, stream>>>(
        ob, wo_b, bo, TOK, DD, DD, 1,
        attn_f, x, nullptr, nullptr, nullptr, nullptr);
    // LN2 -> bf16
    ln_bf16<<<TOK, 256, 0, stream>>>(attn_f, g2, b2, h2);
    // MLP up + exact GELU -> bf16
    gemm_bf16<<<dim3(DFF / 128, TOK / 128), 256, 0, stream>>>(
        h2, wup_b, bup, TOK, DFF, DD, 2,
        nullptr, nullptr, nullptr, nullptr, nullptr, hid);
    // MLP down + residual -> fp32 output
    gemm_bf16<<<dim3(DD / 128, TOK / 128), 256, 0, stream>>>(
        hid, wdn_b, bdown, TOK, DD, DFF, 3,
        out, attn_f, nullptr, nullptr, nullptr, nullptr);
}
